// LaserModel_50835232916065
// MI455X (gfx1250) — matrix operand order repaired and checked by reading, not compiled, not run on hardware
//
#include <hip/hip_runtime.h>
#include <hip/hip_bf16.h>

typedef _Float16 half_t;
typedef __attribute__((ext_vector_type(16))) _Float16 v16h;
typedef __attribute__((ext_vector_type(8)))  _Float16 v8h;
typedef __attribute__((ext_vector_type(8)))  float    v8f;
typedef __attribute__((ext_vector_type(4)))  float    v4f;

#define BSZ 4096
#define TT  256
#define FF  128
#define HH  120   // true hidden size
#define HP  128   // padded hidden size (multiple of 16)

__global__ __launch_bounds__(256, 1)
void rnn_wmma_kernel(const float* __restrict__ x,
                     const float* __restrict__ Wx,
                     const float* __restrict__ Wh,
                     const float* __restrict__ b,
                     const float* __restrict__ Wd,
                     const float* __restrict__ bd,
                     const float* __restrict__ Wo,
                     const float* __restrict__ bo,
                     float* __restrict__ out)
{
    // LDS: double-buffered f16 tiles of x_t and h_t, plus f32 dense output tile
    __shared__ __attribute__((aligned(16))) half_t xbuf[2][16][HP]; // 8 KB
    __shared__ __attribute__((aligned(16))) half_t hbuf[2][16][HP]; // 8 KB
    __shared__ __attribute__((aligned(16))) float  dbuf[16][HP];    // 8 KB

    const int tid  = threadIdx.x;
    const int wave = tid >> 5;      // 0..7 -> N tile of padded H
    const int lane = tid & 31;
    const int ln   = lane & 15;     // C/D: column within tile; B: column
    const int lg   = lane >> 4;     // lane group (0 or 1)
    const int n0   = wave * 16;     // this wave's output-column base
    const int m0   = blockIdx.x * 16; // batch-row base

    // ---- B-fragment builder: W is f32 row-major [kmax_rows x HH(=120)],
    // zero-pad cols >= 120 and rows >= kmax. 32x16 f16 B layout:
    // lane l -> column n0+(l&15); element e -> K = k0 + 16*(l>>4) + e.
    auto load_bfrag = [&](const float* __restrict__ W, int k0, int kmax) -> v16h {
        v16h r;
        const int n  = n0 + ln;
        const int kb = k0 + lg * 16;
        #pragma unroll
        for (int e = 0; e < 16; ++e) {
            const int k = kb + e;
            const float v = (n < HH && k < kmax) ? W[k * HH + n] : 0.0f;
            r[e] = (half_t)v;
        }
        return r;
    };

    // ---- A-fragment from LDS row-major [16][HP] f16 at K-offset k0.
    // 16x32 f16 A layout: lane l -> row l&15; lanes 0-15 hold K k0+{0..15},
    // lanes 16-31 hold K k0+{16..31}. Two ds_load_b128 per fragment.
    auto load_afrag = [&](const half_t (*buf)[HP], int k0) -> v16h {
        const v8h lo = *(const v8h*)&buf[ln][k0 + lg * 16];
        const v8h hi = *(const v8h*)&buf[ln][k0 + lg * 16 + 8];
        v16h r;
        #pragma unroll
        for (int e = 0; e < 8; ++e) { r[e] = lo[e]; r[8 + e] = hi[e]; }
        return r;
    };

    // ---- Build recurrent weight fragments once (held in VGPRs for all 256 steps)
    v16h bxf[4], bhf[4];
    #pragma unroll
    for (int kt = 0; kt < 4; ++kt) {
        bxf[kt] = load_bfrag(Wx, kt * 32, FF);  // Wx: [128 x 120], all rows valid
        bhf[kt] = load_bfrag(Wh, kt * 32, HH);  // Wh: [120 x 120], pad rows >= 120
    }
    const float bias = (n0 + ln < HH) ? b[n0 + ln] : 0.0f;

    // ---- Prologue: zero h_0, stage x_0 (f32->f16) into LDS
    {
        const int i = tid >> 4;         // row 0..15 of the batch tile
        const int c = (tid & 15) * 8;   // 8-element column chunk
        #pragma unroll
        for (int e = 0; e < 8; ++e) hbuf[0][i][c + e] = (half_t)0.0f;
        const float* src = x + ((size_t)(m0 + i) * TT + 0) * FF + c;
        const v4f a0 = *(const v4f*)src;
        const v4f a1 = *(const v4f*)(src + 4);
        v8h hx;
        #pragma unroll
        for (int e = 0; e < 4; ++e) { hx[e] = (half_t)a0[e]; hx[4 + e] = (half_t)a1[e]; }
        *(v8h*)&xbuf[0][i][c] = hx;
    }
    __syncthreads();

    // ---- Time recurrence
    const int pi = tid >> 4;
    const int pc = (tid & 15) * 8;
    for (int t = 0; t < TT; ++t) {
        const int p = t & 1;

        // Prefetch x_{t+1} from HBM while WMMAs run
        v4f pre0{}, pre1{};
        const bool pref = (t + 1 < TT);
        if (pref) {
            const float* src = x + ((size_t)(m0 + pi) * TT + (t + 1)) * FF + pc;
            pre0 = *(const v4f*)src;
            pre1 = *(const v4f*)(src + 4);
        }

        // Two independent accumulator chains (x-path carries the bias)
        v8f accX, accH;
        #pragma unroll
        for (int r = 0; r < 8; ++r) { accX[r] = bias; accH[r] = 0.0f; }

        #pragma unroll
        for (int kt = 0; kt < 4; ++kt) {
            const v16h ax = load_afrag(xbuf[p], kt * 32);
            const v16h ah = load_afrag(hbuf[p], kt * 32);
            accX = __builtin_amdgcn_wmma_f32_16x16x32_f16(false, ax, false, bxf[kt],
                                                          (short)0, accX, false, false);
            accH = __builtin_amdgcn_wmma_f32_16x16x32_f16(false, ah, false, bhf[kt],
                                                          (short)0, accH, false, false);
        }

        // relu + store h_t (f16) to alternate buffer.
        // C/D layout: lane -> col n0+ln; VGPR r -> row r + 8*lg.
        #pragma unroll
        for (int r = 0; r < 8; ++r) {
            const float v = __builtin_fmaxf(accX[r] + accH[r], 0.0f);
            hbuf[1 - p][r + lg * 8][n0 + ln] = (half_t)v;
        }

        if (pref) {
            v8h hx;
            #pragma unroll
            for (int e = 0; e < 4; ++e) { hx[e] = (half_t)pre0[e]; hx[4 + e] = (half_t)pre1[e]; }
            *(v8h*)&xbuf[1 - p][pi][pc] = hx;
        }
        __syncthreads();
    }

    // ---- Dense(120, relu): d = relu(h @ Wd + bd), f32 result tile in LDS
    const half_t (*hf)[HP] = hbuf[TT & 1];
    v16h bdf[4];
    #pragma unroll
    for (int kt = 0; kt < 4; ++kt) bdf[kt] = load_bfrag(Wd, kt * 32, HH);
    const float biasd = (n0 + ln < HH) ? bd[n0 + ln] : 0.0f;

    v8f acc;
    #pragma unroll
    for (int r = 0; r < 8; ++r) acc[r] = biasd;
    #pragma unroll
    for (int kt = 0; kt < 4; ++kt) {
        const v16h a = load_afrag(hf, kt * 32);
        acc = __builtin_amdgcn_wmma_f32_16x16x32_f16(false, a, false, bdf[kt],
                                                     (short)0, acc, false, false);
    }
    #pragma unroll
    for (int r = 0; r < 8; ++r)
        dbuf[r + lg * 8][n0 + ln] = __builtin_fmaxf(acc[r], 0.0f);
    __syncthreads();

    // ---- Dense(1): out[m] = d[m,:] @ Wo + bo  (tiny, plain VALU)
    if (tid < 16) {
        float s = bo[0];
        #pragma unroll 4
        for (int c = 0; c < HH; ++c) s += dbuf[tid][c] * Wo[c];
        out[m0 + tid] = s;
    }
}

extern "C" void kernel_launch(void* const* d_in, const int* in_sizes, int n_in,
                              void* d_out, int out_size, void* d_ws, size_t ws_size,
                              hipStream_t stream) {
    const float* x  = (const float*)d_in[0];
    const float* Wx = (const float*)d_in[1];
    const float* Wh = (const float*)d_in[2];
    const float* b  = (const float*)d_in[3];
    const float* Wd = (const float*)d_in[4];
    const float* bd = (const float*)d_in[5];
    const float* Wo = (const float*)d_in[6];
    const float* bo = (const float*)d_in[7];
    float* out = (float*)d_out;

    dim3 grid(BSZ / 16);   // 256 workgroups, one 16-row batch tile each
    dim3 block(256);       // 8 waves: one per 16-wide slice of padded H
    rnn_wmma_kernel<<<grid, block, 0, stream>>>(x, Wx, Wh, b, Wd, bd, Wo, bo, out);
}
